// GNN_52682068852837
// MI455X (gfx1250) — matrix operand order repaired and checked by reading, not compiled, not run on hardware
//
#include <hip/hip_runtime.h>
#include <hip/hip_fp16.h>

// ---------------------------------------------------------------------------
// GraphNet on MI455X / gfx1250 (wave32):
//  - f16 node features end-to-end (halves HBM gather traffic, feeds WMMA)
//  - gather: global_load_async_to_lds_b64 straight into per-wave LDS act tiles
//  - weights pre-transposed to N-major f16 in workspace; per-block load via
//    Tensor Data Mover (tensor_load_to_lds) with vector-copy fallback
//  - GEMMs: v_wmma_f32_16x16x32_f16, 16-row tiles, LN/relu on accumulators
//  - segment-sum via packed global_atomic_pk_add_f16
// ---------------------------------------------------------------------------

#define NN_NODES 100000
#define NN_EDGES 500000
#define BDIM 128  // 4 waves
#define WAVES 4

typedef __attribute__((ext_vector_type(16))) _Float16 v16h;
typedef __attribute__((ext_vector_type(8))) float v8f;
typedef unsigned int v4u __attribute__((ext_vector_type(4)));
typedef int v8i __attribute__((ext_vector_type(8)));
typedef int v4i __attribute__((ext_vector_type(4)));

struct Mlp4 {
    const float* W[4];
    const float* b[4];
    const float* g[4];
    const float* bt[4];
};

// ---- WMMA fragment loaders (CDNA5 ISA §7.12.2 layouts) ----------

__device__ __forceinline__ v16h load_a_tile(const _Float16* act, int m, int half,
                                            int kBase) {
    v16h a;
#pragma unroll
    for (int j = 0; j < 16; ++j) {
        int v = j >> 1, p = j & 1;
        int k = ((v & 4) ? 16 : 0) + half * 8 + (v & 3) * 2 + p;
        a[j] = act[m * 128 + kBase + k];
    }
    return a;
}

template <int K>
__device__ __forceinline__ v16h load_b_tile(const _Float16* wT, int n, int half,
                                            int kBase) {
    v16h b;
#pragma unroll
    for (int j = 0; j < 16; ++j) b[j] = wT[n * K + kBase + half * 8 + (j >> 3) * 16 + (j & 7)];
    return b;
}

// out(16x64) = relu(LN(act(16xK) @ W + b)); acc in WMMA C/D layout.
template <int K, bool LN, bool WRITE>
__device__ __forceinline__ void wmma_layer(_Float16* act, const _Float16* wT,
                                           const float* bias, const float* gam,
                                           const float* bet, int lane,
                                           v8f acc[4]) {
    const int nl = lane & 15;
    const int half = lane >> 4;
#pragma unroll
    for (int t = 0; t < 4; ++t) {
        float bv = bias[t * 16 + nl];
#pragma unroll
        for (int r = 0; r < 8; ++r) acc[t][r] = bv;
    }
#pragma unroll
    for (int kb = 0; kb < K; kb += 32) {
        v16h a = load_a_tile(act, nl, half, kb);
#pragma unroll
        for (int t = 0; t < 4; ++t) {
            v16h b = load_b_tile<K>(wT, t * 16 + nl, half, kb);
            acc[t] = __builtin_amdgcn_wmma_f32_16x16x32_f16(
                false, a, false, b, (short)0, acc[t], false, false);
        }
    }
    __builtin_amdgcn_wave_barrier();
#pragma unroll
    for (int r = 0; r < 8; ++r) {
        float mean = 0.f, rstd = 1.f;
        if constexpr (LN) {
            float s = 0.f, q = 0.f;
#pragma unroll
            for (int t = 0; t < 4; ++t) {
                float x = acc[t][r];
                s += x;
                q += x * x;
            }
#pragma unroll
            for (int m = 8; m >= 1; m >>= 1) {
                s += __shfl_xor(s, m, 16);
                q += __shfl_xor(q, m, 16);
            }
            mean = s * (1.f / 64.f);
            float var = q * (1.f / 64.f) - mean * mean;
            rstd = rsqrtf(var + 1e-5f);
        }
#pragma unroll
        for (int t = 0; t < 4; ++t) {
            float x = acc[t][r];
            if constexpr (LN)
                x = (x - mean) * rstd * gam[t * 16 + nl] + bet[t * 16 + nl];
            x = fmaxf(x, 0.f);
            acc[t][r] = x;
            if constexpr (WRITE)
                act[(r + half * 8) * 128 + t * 16 + nl] = (_Float16)x;
        }
    }
    __builtin_amdgcn_wave_barrier();
}

// ---- Weight preload: TDM 1-D descriptor, fallback vector copy --------------

__device__ __forceinline__ void load_weights_lds(_Float16* wT,
                                                 const _Float16* wG,
                                                 unsigned nHalves, int tid) {
#if __has_builtin(__builtin_amdgcn_tensor_load_to_lds)
    if (tid < 32) {  // wave 0 issues one TDM op for the whole weight block
        unsigned long long ga = (unsigned long long)(uintptr_t)wG;
        unsigned ldsA = (unsigned)(uintptr_t)wT;  // flat LDS addr: low 32 bits
        v4u g0;
        g0[0] = 1u;  // count=1, user descriptor
        g0[1] = ldsA;
        g0[2] = (unsigned)(ga & 0xFFFFFFFFu);
        g0[3] = (unsigned)((ga >> 32) & 0x01FFFFFFu) | (2u << 30);  // type=2
        v8i g1;
        g1[0] = (1 << 16);  // workgroup_mask=0, data_size=1 (2-byte elements)
        g1[1] = (int)((nHalves & 0xFFFFu) << 16);       // tensor_dim0 lo16
        g1[2] = (int)(((nHalves >> 16) & 0xFFFFu) | (1u << 16));  // dim0 hi, dim1=1
        g1[3] = (int)((nHalves & 0xFFFFu) << 16);       // tile_dim0
        g1[4] = 0;                                      // tile_dim1/2 unused
        g1[5] = (int)nHalves;                           // dim0 stride lo32
        g1[6] = 0;
        g1[7] = 0;
        v4i gz = {0, 0, 0, 0};
#if __clang_major__ >= 23
        v8i gz8 = {0, 0, 0, 0, 0, 0, 0, 0};
        __builtin_amdgcn_tensor_load_to_lds(g0, g1, gz, gz, gz8, 0);
#else
        __builtin_amdgcn_tensor_load_to_lds(g0, g1, gz, gz, 0);
#endif
#if __has_builtin(__builtin_amdgcn_s_wait_tensorcnt)
        __builtin_amdgcn_s_wait_tensorcnt(0);
#else
        asm volatile("s_wait_tensorcnt 0x0" ::: "memory");
#endif
    }
#else
    uint4* d = (uint4*)wT;
    const uint4* s = (const uint4*)wG;
    for (unsigned i = tid; i < nHalves / 8u; i += BDIM) d[i] = s[i];
#endif
}

__device__ __forceinline__ void load_f32(float* dst, const float* src, int n,
                                         int tid) {
    for (int i = tid; i < n; i += BDIM) dst[i] = src[i];
}

// ---- async gather of one 16x128 f16 act tile -------------------------------
// row e: lanes 0-15 pull 8B chunks of rowA, lanes 16-31 of rowB.
__device__ __forceinline__ void async_row(unsigned actBase, int e,
                                          const _Float16* src) {
    unsigned lds = actBase + (unsigned)(e * 256) + (unsigned)(threadIdx.x & 31) * 8u;
    asm volatile("global_load_async_to_lds_b64 %0, %1, off" ::"v"(lds),
                 "v"((unsigned long long)(uintptr_t)src)
                 : "memory");
}

// ---- Kernels ---------------------------------------------------------------

__global__ void zero_kernel(unsigned* __restrict__ p, int n) {
    int i = blockIdx.x * blockDim.x + threadIdx.x;
    if (i < n) p[i] = 0u;
}

// transpose+convert W [K][64] f32 -> [64][K] f16
__global__ void prep_wT(_Float16* __restrict__ dst, const float* __restrict__ src,
                        int K) {
    int i = blockIdx.x * blockDim.x + threadIdx.x;
    if (i < K * 64) {
        int k = i >> 6, n = i & 63;
        dst[n * K + k] = (_Float16)src[i];
    }
}

// h = relu(relu(x @ W0 + b0) @ W1 + b1), f16 output
__global__ __launch_bounds__(BDIM) void enc_kernel(
    const float* __restrict__ x, const float* __restrict__ W0,
    const float* __restrict__ W1, const float* __restrict__ b0,
    const float* __restrict__ b1, _Float16* __restrict__ h, int nNodes) {
    __shared__ float w0[3 * 64];
    __shared__ float w1[64 * 64];
    __shared__ float bb0[64], bb1[64];
    int tid = threadIdx.x;
    for (int i = tid; i < 192; i += BDIM) w0[i] = W0[i];
    for (int i = tid; i < 4096; i += BDIM) w1[i] = W1[i];
    for (int i = tid; i < 64; i += BDIM) {
        bb0[i] = b0[i];
        bb1[i] = b1[i];
    }
    __syncthreads();
    int n = blockIdx.x * BDIM + tid;
    float x0 = 0.f, x1 = 0.f, x2 = 0.f;
    if (n < nNodes) {
        x0 = x[(size_t)n * 3 + 0];
        x1 = x[(size_t)n * 3 + 1];
        x2 = x[(size_t)n * 3 + 2];
    }
    float t1[64];
#pragma unroll
    for (int f = 0; f < 64; ++f)
        t1[f] = fmaxf(
            fmaf(x2, w0[128 + f], fmaf(x1, w0[64 + f], fmaf(x0, w0[f], bb0[f]))),
            0.f);
    if (n < nNodes) {
        for (int f = 0; f < 64; ++f) {
            float s = bb1[f];
#pragma unroll 8
            for (int k = 0; k < 64; ++k) s = fmaf(t1[k], w1[k * 64 + f], s);
            h[(size_t)n * 64 + f] = (_Float16)fmaxf(s, 0.f);
        }
    }
}

// Fused: async gather [h[send]|h[recv]] -> 4-layer LN-MLP -> pk-f16 atomics.
template <int TILES>
__global__ __launch_bounds__(BDIM) void edge_kernel(
    const _Float16* __restrict__ h, const int* __restrict__ ei,
    const _Float16* __restrict__ wG, Mlp4 p, _Float16* __restrict__ aggr,
    int nE, int totalRows) {
    __shared__ _Float16 wT[20480];
    __shared__ float bias[256], gam[256], bet[256];
    __shared__ _Float16 act_s[WAVES][2048];

    int tid = threadIdx.x;
    load_weights_lds(wT, wG, 20480u, tid);
#pragma unroll
    for (int l = 0; l < 4; ++l) {
        load_f32(bias + l * 64, p.b[l], 64, tid);
        load_f32(gam + l * 64, p.g[l], 64, tid);
        load_f32(bet + l * 64, p.bt[l], 64, tid);
    }
    __syncthreads();

    int wave = tid >> 5, lane = tid & 31;
    int nl = lane & 15, half = lane >> 4;
    _Float16* act = act_s[wave];
    unsigned actB = (unsigned)(uintptr_t)act;

    for (int tt = 0; tt < TILES; ++tt) {
        int rowBase = ((blockIdx.x * WAVES + wave) * TILES + tt) * 16;
        int sIdx = 0, rIdx = 0;
        int myRow = rowBase + nl;
        int rowC = (myRow < totalRows) ? myRow : 0;
        if (lane < 16) {
            sIdx = ei[rowC];
            rIdx = (rowC < nE) ? ei[nE + rowC] : ei[rowC - nE];
        }

        asm volatile("s_wait_dscnt 0x0" ::: "memory");  // act reads done
#pragma unroll 4
        for (int e = 0; e < 16; ++e) {
            int s = __shfl(sIdx, e, 32);
            int rr = __shfl(rIdx, e, 32);
            async_row(actB, e, h + (size_t)(half ? rr : s) * 64 + nl * 4);
        }
        asm volatile("s_wait_asynccnt 0x0" ::: "memory");

        v8f acc[4];
        wmma_layer<128, true, true>(act, wT, bias, gam, bet, lane, acc);
        wmma_layer<64, true, true>(act, wT + 8192, bias + 64, gam + 64,
                                   bet + 64, lane, acc);
        wmma_layer<64, true, true>(act, wT + 12288, bias + 128, gam + 128,
                                   bet + 128, lane, acc);
        wmma_layer<64, true, false>(act, wT + 16384, bias + 192, gam + 192,
                                    bet + 192, lane, acc);

        // segment-sum: packed f16 atomics (even lanes carry [f, f+1])
#pragma unroll
        for (int r = 0; r < 8; ++r) {
            int row = r + half * 8;
            int gr = rowBase + row;
            int rv = __shfl(rIdx, row, 32);
#pragma unroll
            for (int t = 0; t < 4; ++t) {
                float x = acc[t][r];
                float xp = __shfl_xor(x, 1, 32);
                if (gr < totalRows && (nl & 1) == 0) {
                    __half2 v = __floats2half2_rn(x, xp);
                    atomicAdd((__half2*)(aggr + (size_t)rv * 64 + t * 16 + nl),
                              v);
                }
            }
        }
    }
}

// Fused: async gather [h|aggr] -> 4-layer LN-MLP -> +h residual -> hout (f16)
template <int TILES>
__global__ __launch_bounds__(BDIM) void node_kernel(
    const _Float16* __restrict__ h, const _Float16* __restrict__ aggr,
    const _Float16* __restrict__ wG, Mlp4 p, _Float16* __restrict__ hout,
    int nNodes) {
    __shared__ _Float16 wT[20480];
    __shared__ float bias[256], gam[256], bet[256];
    __shared__ _Float16 act_s[WAVES][2048];

    int tid = threadIdx.x;
    load_weights_lds(wT, wG, 20480u, tid);
#pragma unroll
    for (int l = 0; l < 4; ++l) {
        load_f32(bias + l * 64, p.b[l], 64, tid);
        load_f32(gam + l * 64, p.g[l], 64, tid);
        load_f32(bet + l * 64, p.bt[l], 64, tid);
    }
    __syncthreads();

    int wave = tid >> 5, lane = tid & 31;
    int nl = lane & 15, half = lane >> 4;
    _Float16* act = act_s[wave];
    unsigned actB = (unsigned)(uintptr_t)act;

    for (int tt = 0; tt < TILES; ++tt) {
        int rowBase = ((blockIdx.x * WAVES + wave) * TILES + tt) * 16;

        asm volatile("s_wait_dscnt 0x0" ::: "memory");
#pragma unroll 4
        for (int e = 0; e < 16; ++e) {
            int node = rowBase + e;
            if (node >= nNodes) node = 0;
            const _Float16* src =
                (half ? aggr : h) + (size_t)node * 64 + nl * 4;
            async_row(actB, e, src);
        }
        asm volatile("s_wait_asynccnt 0x0" ::: "memory");

        v8f acc[4];
        wmma_layer<128, true, true>(act, wT, bias, gam, bet, lane, acc);
        wmma_layer<64, true, true>(act, wT + 8192, bias + 64, gam + 64,
                                   bet + 64, lane, acc);
        wmma_layer<64, true, true>(act, wT + 12288, bias + 128, gam + 128,
                                   bet + 128, lane, acc);
        wmma_layer<64, true, false>(act, wT + 16384, bias + 192, gam + 192,
                                    bet + 192, lane, acc);

#pragma unroll
        for (int r = 0; r < 8; ++r) {
            int gn = rowBase + r + half * 8;
            if (gn < nNodes) {
#pragma unroll
                for (int t = 0; t < 4; ++t) {
                    int f = t * 16 + nl;
                    float h0 = (float)h[(size_t)gn * 64 + f];
                    hout[(size_t)gn * 64 + f] = (_Float16)(acc[t][r] + h0);
                }
            }
        }
    }
}

// Classifier: async gather [h[s]|h[t]] -> relu(128->64) -> dot(64->1)
template <int TILES>
__global__ __launch_bounds__(BDIM) void clf_kernel(
    const _Float16* __restrict__ h, const int* __restrict__ ei,
    const _Float16* __restrict__ wG, const float* __restrict__ b0,
    const float* __restrict__ W1, const float* __restrict__ b1,
    float* __restrict__ out, int nE) {
    __shared__ _Float16 wT[8192];
    __shared__ float bias[64], w2[64];
    __shared__ float b2s;
    __shared__ _Float16 act_s[WAVES][2048];

    int tid = threadIdx.x;
    load_weights_lds(wT, wG, 8192u, tid);
    load_f32(bias, b0, 64, tid);
    load_f32(w2, W1, 64, tid);
    if (tid == 0) b2s = b1[0];
    __syncthreads();

    int wave = tid >> 5, lane = tid & 31;
    int nl = lane & 15, half = lane >> 4;
    _Float16* act = act_s[wave];
    unsigned actB = (unsigned)(uintptr_t)act;

    for (int tt = 0; tt < TILES; ++tt) {
        int rowBase = ((blockIdx.x * WAVES + wave) * TILES + tt) * 16;
        int sIdx = 0, tIdx = 0;
        int myRow = rowBase + nl;
        int rowC = (myRow < nE) ? myRow : 0;
        if (lane < 16) {
            sIdx = ei[rowC];
            tIdx = ei[nE + rowC];
        }

        asm volatile("s_wait_dscnt 0x0" ::: "memory");
#pragma unroll 4
        for (int e = 0; e < 16; ++e) {
            int s = __shfl(sIdx, e, 32);
            int tt2 = __shfl(tIdx, e, 32);
            async_row(actB, e, h + (size_t)(half ? tt2 : s) * 64 + nl * 4);
        }
        asm volatile("s_wait_asynccnt 0x0" ::: "memory");

        v8f acc[4];
        wmma_layer<128, false, false>(act, wT, bias, nullptr, nullptr, lane,
                                      acc);

#pragma unroll
        for (int r = 0; r < 8; ++r) {
            float s = acc[0][r] * w2[nl] + acc[1][r] * w2[16 + nl] +
                      acc[2][r] * w2[32 + nl] + acc[3][r] * w2[48 + nl];
#pragma unroll
            for (int m = 8; m >= 1; m >>= 1) s += __shfl_xor(s, m, 16);
            int gr = rowBase + r + half * 8;
            if (nl == 0 && gr < nE) out[gr] = s + b2s;
        }
    }
}

// ---------------------------------------------------------------------------
// d_in order: x, edge_index, then params pytree (JAX dict keys sorted):
// clf{W0,W1,b0,b1}, edge{W0..3,b0..3,bt0..3,g0..3}, enc{W0,W1,b0,b1},
// node{W0..3,b0..3,bt0..3,g0..3}.
// ---------------------------------------------------------------------------
extern "C" void kernel_launch(void* const* d_in, const int* in_sizes, int n_in,
                              void* d_out, int out_size, void* d_ws,
                              size_t ws_size, hipStream_t stream) {
    const float* x = (const float*)d_in[0];
    const int* ei = (const int*)d_in[1];
    int i = 2;
    const float* clfW0 = (const float*)d_in[i++];
    const float* clfW1 = (const float*)d_in[i++];
    const float* clfb0 = (const float*)d_in[i++];
    const float* clfb1 = (const float*)d_in[i++];
    Mlp4 edgeP, nodeP;
    for (int l = 0; l < 4; ++l) edgeP.W[l] = (const float*)d_in[i++];
    for (int l = 0; l < 4; ++l) edgeP.b[l] = (const float*)d_in[i++];
    for (int l = 0; l < 4; ++l) edgeP.bt[l] = (const float*)d_in[i++];
    for (int l = 0; l < 4; ++l) edgeP.g[l] = (const float*)d_in[i++];
    const float* encW0 = (const float*)d_in[i++];
    const float* encW1 = (const float*)d_in[i++];
    const float* encb0 = (const float*)d_in[i++];
    const float* encb1 = (const float*)d_in[i++];
    for (int l = 0; l < 4; ++l) nodeP.W[l] = (const float*)d_in[i++];
    for (int l = 0; l < 4; ++l) nodeP.b[l] = (const float*)d_in[i++];
    for (int l = 0; l < 4; ++l) nodeP.bt[l] = (const float*)d_in[i++];
    for (int l = 0; l < 4; ++l) nodeP.g[l] = (const float*)d_in[i++];

    _Float16* hA = (_Float16*)d_ws;                   // N*64 f16
    _Float16* hB = hA + (size_t)NN_NODES * 64;        // N*64 f16
    _Float16* aggr = hB + (size_t)NN_NODES * 64;      // N*64 f16
    _Float16* wEdge = aggr + (size_t)NN_NODES * 64;   // 20480 f16, N-major
    _Float16* wNode = wEdge + 20480;                  // 20480 f16
    _Float16* wClf = wNode + 20480;                   // 8192 f16

    // one-shot: transpose+convert weights to N-major f16
    prep_wT<<<(128 * 64 + 255) / 256, 256, 0, stream>>>(wEdge, edgeP.W[0], 128);
    prep_wT<<<(64 * 64 + 255) / 256, 256, 0, stream>>>(wEdge + 8192, edgeP.W[1], 64);
    prep_wT<<<(64 * 64 + 255) / 256, 256, 0, stream>>>(wEdge + 12288, edgeP.W[2], 64);
    prep_wT<<<(64 * 64 + 255) / 256, 256, 0, stream>>>(wEdge + 16384, edgeP.W[3], 64);
    prep_wT<<<(128 * 64 + 255) / 256, 256, 0, stream>>>(wNode, nodeP.W[0], 128);
    prep_wT<<<(64 * 64 + 255) / 256, 256, 0, stream>>>(wNode + 8192, nodeP.W[1], 64);
    prep_wT<<<(64 * 64 + 255) / 256, 256, 0, stream>>>(wNode + 12288, nodeP.W[2], 64);
    prep_wT<<<(64 * 64 + 255) / 256, 256, 0, stream>>>(wNode + 16384, nodeP.W[3], 64);
    prep_wT<<<(128 * 64 + 255) / 256, 256, 0, stream>>>(wClf, clfW0, 128);

    enc_kernel<<<(NN_NODES + BDIM - 1) / BDIM, BDIM, 0, stream>>>(
        x, encW0, encW1, encb0, encb1, hA, NN_NODES);

    _Float16* hc = hA;
    _Float16* hn = hB;
    const int totalRows = 2 * NN_EDGES;
    const int eTiles = (totalRows + 15) / 16;
    const int nTiles = (NN_NODES + 15) / 16;
    const int cTiles = (NN_EDGES + 15) / 16;
    const int eBlocks = (eTiles + WAVES * 4 - 1) / (WAVES * 4);
    const int nBlocks = (nTiles + WAVES * 2 - 1) / (WAVES * 2);
    const int cBlocks = (cTiles + WAVES * 4 - 1) / (WAVES * 4);

    for (int it = 0; it < 2; ++it) {
        zero_kernel<<<(NN_NODES * 32 + 255) / 256, 256, 0, stream>>>(
            (unsigned*)aggr, NN_NODES * 32);
        edge_kernel<4><<<eBlocks, BDIM, 0, stream>>>(hc, ei, wEdge, edgeP,
                                                     aggr, NN_EDGES, totalRows);
        node_kernel<2><<<nBlocks, BDIM, 0, stream>>>(hc, aggr, wNode, nodeP,
                                                     hn, NN_NODES);
        _Float16* tmp = hc;
        hc = hn;
        hn = tmp;
    }

    clf_kernel<4><<<cBlocks, BDIM, 0, stream>>>(hc, ei, wClf, clfb0, clfW1,
                                                clfb1, (float*)d_out, NN_EDGES);
}
